// SwinTransformerBlock_82858509074997
// MI455X (gfx1250) — matrix operand order repaired and checked by reading, not compiled, not run on hardware
//
#include <hip/hip_runtime.h>
#include <hip/hip_bf16.h>

// ---------------------------------------------------------------------------
// Swin-3D transformer block for MI455X (gfx1250, wave32, WMMA f16->f32).
// All matmuls via v_wmma_f32_16x16x32_f16; fp16 intermediates to minimize
// HBM traffic. GEMM: 128x128x32 block tile, 8 WMMAs/wave/K-step,
// double-buffered LDS, one barrier per K-step, async global->LDS DMA
// (ASYNCcnt path) with register-staging fallback.
// ---------------------------------------------------------------------------

typedef __attribute__((ext_vector_type(16))) _Float16 v16h;
typedef __attribute__((ext_vector_type(8)))  _Float16 v8h;
typedef __attribute__((ext_vector_type(8)))  float    v8f;
typedef __attribute__((ext_vector_type(4)))  int      v4i;

#define DIM       384
#define HEADS     12
#define HD        32
#define NTOK      64
#define NWIN      1024
#define NROWS     65536          // total tokens
#define QKV_N     1152
#define MLP_H     1536
#define TABLE     343

#if __has_builtin(__builtin_amdgcn_global_load_async_to_lds_b128)
#define HAVE_ASYNC_LDS 1
#else
#define HAVE_ASYNC_LDS 0
#endif

// ---------------- WMMA / async helpers ------------------------------------

static __device__ __forceinline__ v8f wmma_f16(v16h a, v16h b, v8f c) {
    return __builtin_amdgcn_wmma_f32_16x16x32_f16(
        /*neg_a=*/false, a, /*neg_b=*/false, b,
        /*c_mod=*/(short)0, c, /*reuse_a=*/false, /*reuse_b=*/false);
}

#if HAVE_ASYNC_LDS
static __device__ __forceinline__ void async_cp16(const _Float16* g, _Float16* l) {
    __builtin_amdgcn_global_load_async_to_lds_b128(
        (__attribute__((address_space(1))) v4i*)g,
        (__attribute__((address_space(3))) v4i*)l,
        0, 0);
}
static __device__ __forceinline__ void wait_async0() {
#if __has_builtin(__builtin_amdgcn_s_wait_asynccnt)
    __builtin_amdgcn_s_wait_asynccnt(0);
#else
    asm volatile("s_wait_asynccnt 0" ::: "memory");
#endif
}
#endif

static __device__ __forceinline__ v16h cat8(v8h lo, v8h hi) {
    v16h r;
#pragma unroll
    for (int i = 0; i < 8; ++i) { r[i] = lo[i]; r[8 + i] = hi[i]; }
    return r;
}

// A-fragment (16x32 f16, MxK). ISA layout: lane<16 holds M=lane, K {0..7,16..23};
// lane>=16 holds M=lane-16, K {8..15,24..31}.
static __device__ __forceinline__ v16h frag_a_lds(const _Float16* tile, int ld) {
    const int lane = threadIdx.x & 31;
    const int m  = lane & 15;
    const int kb = (lane >> 4) << 3;             // 0 or 8
    const _Float16* p = tile + m * ld + kb;
    v8h lo = *(const v8h*)p;                     // K = kb..kb+7
    v8h hi = *(const v8h*)(p + 16);              // K = kb+16..kb+23
    return cat8(lo, hi);
}

// B-fragment (32x16 f16, KxN) sourced from B^T (row n holds K contiguous).
// lane<16: col=lane, K {0..7,16..23} ; lane>=16: col=lane-16, K {8..15,24..31}.
static __device__ __forceinline__ v16h frag_b_lds(const _Float16* tileT, int ld) {
    const int lane = threadIdx.x & 31;
    const int n  = lane & 15;
    const int kb = (lane >> 4) << 3;             // 0 or 8
    const _Float16* p = tileT + n * ld + kb;
    v8h lo = *(const v8h*)p;
    v8h hi = *(const v8h*)(p + 16);
    return cat8(lo, hi);
}

static __device__ __forceinline__ float wave_sum(float v) {
#pragma unroll
    for (int o = 16; o > 0; o >>= 1) v += __shfl_xor(v, o, 32);
    return v;
}

static __device__ __forceinline__ float gelu_exact(float x) {
    return 0.5f * x * (1.0f + erff(x * 0.70710678118654752f));
}

// ---------------- weight convert: fp32 [K][N] -> fp16 B^T [N][K] ----------

__global__ void __launch_bounds__(256)
cvt_t_kernel(const float* __restrict__ W, _Float16* __restrict__ Wt,
             int K, int N) {
    size_t i = (size_t)blockIdx.x * 256 + threadIdx.x;
    if (i >= (size_t)K * N) return;
    int n = (int)(i / K);
    int k = (int)(i % K);
    Wt[i] = (_Float16)W[(size_t)k * N + n];
}

// ---------------- LN1 + shift + window partition --------------------------
// window-ordered row r = win*64+tok ; source = roll(x,-2) at window coords.

__global__ void __launch_bounds__(256)
ln1_win_kernel(const float* __restrict__ x, const float* __restrict__ g,
               const float* __restrict__ b, _Float16* __restrict__ yw) {
    const int row  = blockIdx.x * 8 + (threadIdx.x >> 5);
    const int lane = threadIdx.x & 31;
    const int win = row >> 6, tok = row & 63;
    const int wz = win >> 8, wy = (win >> 4) & 15, wx = win & 15;
    const int tz = tok >> 4, ty = (tok >> 2) & 3, tx = tok & 3;
    const int z  = (wz * 4 + tz + 2) & 15;
    const int y  = (wy * 4 + ty + 2) & 63;
    const int xx = (wx * 4 + tx + 2) & 63;
    const size_t src = ((size_t)(z * 64 + y) * 64 + xx) * DIM;

    float v[12], s = 0.f;
#pragma unroll
    for (int i = 0; i < 12; ++i) { v[i] = x[src + lane + i * 32]; s += v[i]; }
    const float mu = wave_sum(s) * (1.f / DIM);
    float q = 0.f;
#pragma unroll
    for (int i = 0; i < 12; ++i) { float d = v[i] - mu; q += d * d; }
    const float rs = rsqrtf(wave_sum(q) * (1.f / DIM) + 1e-5f);
#pragma unroll
    for (int i = 0; i < 12; ++i) {
        const int c = lane + i * 32;
        yw[(size_t)row * DIM + c] = (_Float16)((v[i] - mu) * rs * g[c] + b[c]);
    }
}

// ---------------- generic WMMA GEMM ---------------------------------------
// C[M,N] = A[M,K] * B[K,N] (+bias). A fp16 row-major, Bt fp16 [N][K].
// Block: 256 thr = 8 waves -> 128x128 tile; wave = 32(M) x 64(N) = 8 WMMA.
// Double-buffered LDS, one barrier per 32-K step.
// MODE 0: fp16 out (bias)   MODE 1: fp16 out (bias+GELU)
// MODE 2: fp32 out += acc+bias (residual accumulate)

#define BMT 128
#define BNT 128
#define BKT 32
#define LDT 40   // padded LDS stride (distinct banks across 16 lanes)

template <int MODE>
__global__ void __launch_bounds__(256)
gemm_wmma(const _Float16* __restrict__ A, const _Float16* __restrict__ Bt,
          const float* __restrict__ bias, _Float16* __restrict__ Of16,
          float* __restrict__ Of32, int M, int N, int K) {
    __shared__ _Float16 As[2][BMT * LDT];
    __shared__ _Float16 Bs[2][BMT * LDT];
    const int bm = blockIdx.y * BMT;
    const int bn = blockIdx.x * BNT;
    const int tid  = threadIdx.x;
    const int wave = tid >> 5, lane = tid & 31;
    const int wm = wave >> 1;          // 0..3 : M offset wm*32
    const int wn = wave & 1;           // 0..1 : N offset wn*64

    v8f acc[2][4];
#pragma unroll
    for (int i = 0; i < 2; ++i)
#pragma unroll
        for (int j = 0; j < 4; ++j) acc[i][j] = (v8f){};

    // loader coords: each thread fills 2 rows x 8 halves per matrix
    const int r0 = tid >> 2;           // 0..63
    const int r1 = r0 + 64;            // 64..127
    const int ca = (tid & 3) << 3;     // 0,8,16,24
    const _Float16* Ar0 = A  + (size_t)(bm + r0) * K + ca;
    const _Float16* Ar1 = A  + (size_t)(bm + r1) * K + ca;
    const _Float16* Br0 = Bt + (size_t)(bn + r0) * K + ca;
    const _Float16* Br1 = Bt + (size_t)(bn + r1) * K + ca;

#if HAVE_ASYNC_LDS
    // ---- async DMA path: global -> LDS directly, tracked with ASYNCcnt ----
    async_cp16(Ar0, As[0] + r0 * LDT + ca);
    async_cp16(Ar1, As[0] + r1 * LDT + ca);
    async_cp16(Br0, Bs[0] + r0 * LDT + ca);
    async_cp16(Br1, Bs[0] + r1 * LDT + ca);
    wait_async0();
    __syncthreads();

    int buf = 0;
    for (int k0 = BKT; k0 < K; k0 += BKT) {
        // DMA next K-slab into the other buffer while we do WMMAs
        async_cp16(Ar0 + k0, As[buf ^ 1] + r0 * LDT + ca);
        async_cp16(Ar1 + k0, As[buf ^ 1] + r1 * LDT + ca);
        async_cp16(Br0 + k0, Bs[buf ^ 1] + r0 * LDT + ca);
        async_cp16(Br1 + k0, Bs[buf ^ 1] + r1 * LDT + ca);

        v16h af0 = frag_a_lds(As[buf] + (wm * 32) * LDT, LDT);
        v16h af1 = frag_a_lds(As[buf] + (wm * 32 + 16) * LDT, LDT);
#pragma unroll
        for (int j = 0; j < 4; ++j) {
            v16h bf = frag_b_lds(Bs[buf] + (wn * 64 + j * 16) * LDT, LDT);
            acc[0][j] = wmma_f16(af0, bf, acc[0][j]);
            acc[1][j] = wmma_f16(af1, bf, acc[1][j]);
        }
        wait_async0();
        __syncthreads();
        buf ^= 1;
    }
#else
    // ---- fallback: stage through registers ----
    {
        v8h a0 = *(const v8h*)Ar0;
        v8h a1 = *(const v8h*)Ar1;
        v8h b0 = *(const v8h*)Br0;
        v8h b1 = *(const v8h*)Br1;
        *(v8h*)(As[0] + r0 * LDT + ca) = a0;
        *(v8h*)(As[0] + r1 * LDT + ca) = a1;
        *(v8h*)(Bs[0] + r0 * LDT + ca) = b0;
        *(v8h*)(Bs[0] + r1 * LDT + ca) = b1;
    }
    __syncthreads();

    int buf = 0;
    for (int k0 = BKT; k0 < K; k0 += BKT) {
        v8h a0 = *(const v8h*)(Ar0 + k0);
        v8h a1 = *(const v8h*)(Ar1 + k0);
        v8h b0 = *(const v8h*)(Br0 + k0);
        v8h b1 = *(const v8h*)(Br1 + k0);

        v16h af0 = frag_a_lds(As[buf] + (wm * 32) * LDT, LDT);
        v16h af1 = frag_a_lds(As[buf] + (wm * 32 + 16) * LDT, LDT);
#pragma unroll
        for (int j = 0; j < 4; ++j) {
            v16h bf = frag_b_lds(Bs[buf] + (wn * 64 + j * 16) * LDT, LDT);
            acc[0][j] = wmma_f16(af0, bf, acc[0][j]);
            acc[1][j] = wmma_f16(af1, bf, acc[1][j]);
        }

        *(v8h*)(As[buf ^ 1] + r0 * LDT + ca) = a0;
        *(v8h*)(As[buf ^ 1] + r1 * LDT + ca) = a1;
        *(v8h*)(Bs[buf ^ 1] + r0 * LDT + ca) = b0;
        *(v8h*)(Bs[buf ^ 1] + r1 * LDT + ca) = b1;
        __syncthreads();
        buf ^= 1;
    }
#endif

    // last K-slab
    {
        v16h af0 = frag_a_lds(As[buf] + (wm * 32) * LDT, LDT);
        v16h af1 = frag_a_lds(As[buf] + (wm * 32 + 16) * LDT, LDT);
#pragma unroll
        for (int j = 0; j < 4; ++j) {
            v16h bf = frag_b_lds(Bs[buf] + (wn * 64 + j * 16) * LDT, LDT);
            acc[0][j] = wmma_f16(af0, bf, acc[0][j]);
            acc[1][j] = wmma_f16(af1, bf, acc[1][j]);
        }
    }

    // epilogue
    const int lane15 = lane & 15;
    const int rgrp   = (lane >> 4) << 3;
#pragma unroll
    for (int j = 0; j < 4; ++j) {
        const int col = bn + wn * 64 + j * 16 + lane15;
        const float bia = bias ? bias[col] : 0.f;
#pragma unroll
        for (int i = 0; i < 2; ++i) {
            const int rbase = bm + wm * 32 + i * 16 + rgrp;
#pragma unroll
            for (int r = 0; r < 8; ++r) {
                float v = acc[i][j][r] + bia;
                if (MODE == 1) v = gelu_exact(v);
                if (MODE == 2) Of32[(size_t)(rbase + r) * N + col] += v;
                else           Of16[(size_t)(rbase + r) * N + col] = (_Float16)v;
            }
        }
    }
}

// ---------------- attention: one block per (window, head) -----------------
// S^T = K * Q^T  (softmax over keys stays within a wave), O = P * V.

__global__ void __launch_bounds__(128)
attn_kernel(const _Float16* __restrict__ qkv, const float* __restrict__ rpb,
            _Float16* __restrict__ out) {
    constexpr int LQ = 40, LV = 72, LP = 72;
    __shared__ _Float16 Qs[64 * LQ];
    __shared__ _Float16 Ks[64 * LQ];
    __shared__ _Float16 Vts[32 * LV];
    __shared__ _Float16 Ps[64 * LP];
    __shared__ float    rpb_s[TABLE];

    const int h    = blockIdx.x;       // head
    const int win  = blockIdx.y;       // window
    const int tid  = threadIdx.x;
    const int wave = tid >> 5, lane = tid & 31;
    const size_t base = (size_t)win * NTOK;

    {   // stage Q, K row-major; V transposed to [c][tok]
        const int t  = tid >> 1;            // token 0..63
        const int c0 = (tid & 1) << 4;      // 0 or 16
        const _Float16* row = qkv + (base + t) * QKV_N + h * HD + c0;
        v8h q0 = *(const v8h*)(row);        v8h q1 = *(const v8h*)(row + 8);
        v8h k0 = *(const v8h*)(row + 384);  v8h k1 = *(const v8h*)(row + 392);
        v8h w0 = *(const v8h*)(row + 768);  v8h w1 = *(const v8h*)(row + 776);
        *(v8h*)(Qs + t * LQ + c0)     = q0; *(v8h*)(Qs + t * LQ + c0 + 8) = q1;
        *(v8h*)(Ks + t * LQ + c0)     = k0; *(v8h*)(Ks + t * LQ + c0 + 8) = k1;
#pragma unroll
        for (int j = 0; j < 8; ++j) {
            Vts[(c0 + j)     * LV + t] = w0[j];
            Vts[(c0 + 8 + j) * LV + t] = w1[j];
        }
    }
    for (int i = tid; i < TABLE; i += 128) rpb_s[i] = rpb[i * HEADS + h];
    __syncthreads();

    // ---- S^T tiles: queries n in [wave*16, +16), all 64 keys (M dim) ----
    v8f sacc[4];
#pragma unroll
    for (int t = 0; t < 4; ++t) sacc[t] = (v8f){};
    v16h bq = frag_b_lds(Qs + (wave * 16) * LQ, LQ);   // B = Q^T
#pragma unroll
    for (int t = 0; t < 4; ++t) {
        v16h ak = frag_a_lds(Ks + (t * 16) * LQ, LQ);  // A = K rows
        sacc[t] = wmma_f16(ak, bq, sacc[t]);
    }

    const int nq   = wave * 16 + (lane & 15);          // query token
    const int koff = (lane >> 4) << 3;                 // key sub-offset
    const int qz = nq >> 4, qy = (nq >> 2) & 3, qx = nq & 3;
    float vals[32];
    float mx = -1e30f;
#pragma unroll
    for (int t = 0; t < 4; ++t) {
#pragma unroll
        for (int r = 0; r < 8; ++r) {
            const int m  = t * 16 + koff + r;          // key token
            const int mz = m >> 4, my = (m >> 2) & 3, mw = m & 3;
            const int idx = (qz - mz + 3) * 49 + (qy - my + 3) * 7 + (qx - mw + 3);
            const float v = sacc[t][r] * 0.17677669529663687f + rpb_s[idx];
            vals[t * 8 + r] = v;
            mx = fmaxf(mx, v);
        }
    }
    mx = fmaxf(mx, __shfl_xor(mx, 16, 32));            // partner lane: other keys
    float se = 0.f;
#pragma unroll
    for (int i = 0; i < 32; ++i) { vals[i] = __expf(vals[i] - mx); se += vals[i]; }
    se += __shfl_xor(se, 16, 32);
    const float inv = 1.f / se;
#pragma unroll
    for (int t = 0; t < 4; ++t)
#pragma unroll
        for (int r = 0; r < 8; ++r)
            Ps[nq * LP + (t * 16 + koff + r)] = (_Float16)(vals[t * 8 + r] * inv);
    __syncthreads();

    // ---- O = P @ V : 16 queries x 32 channels per wave, K(keys)=64 ----
    v8f oa = {}, ob = {};
#pragma unroll
    for (int k0 = 0; k0 < 64; k0 += 32) {
        v16h ap  = frag_a_lds(Ps + (wave * 16) * LP + k0, LP);
        v16h bv0 = frag_b_lds(Vts + 0  * LV + k0, LV);
        v16h bv1 = frag_b_lds(Vts + 16 * LV + k0, LV);
        oa = wmma_f16(ap, bv0, oa);
        ob = wmma_f16(ap, bv1, ob);
    }
    const int cc = lane & 15;
    const int rb = wave * 16 + ((lane >> 4) << 3);
#pragma unroll
    for (int r = 0; r < 8; ++r) {
        const size_t orow = (base + rb + r) * DIM + h * HD;
        out[orow + cc]      = (_Float16)oa[r];
        out[orow + 16 + cc] = (_Float16)ob[r];
    }
}

// ---------------- residual + window-reverse/unshift + LN2 -----------------

__global__ void __launch_bounds__(256)
resid_ln2_kernel(const float* __restrict__ x, const _Float16* __restrict__ proj,
                 const float* __restrict__ g, const float* __restrict__ b,
                 float* __restrict__ outp, _Float16* __restrict__ h1) {
    const int row  = blockIdx.x * 8 + (threadIdx.x >> 5);  // window-order row
    const int lane = threadIdx.x & 31;
    const int win = row >> 6, tok = row & 63;
    const int wz = win >> 8, wy = (win >> 4) & 15, wx = win & 15;
    const int tz = tok >> 4, ty = (tok >> 2) & 3, tx = tok & 3;
    const int z  = (wz * 4 + tz + 2) & 15;
    const int y  = (wy * 4 + ty + 2) & 63;
    const int xx = (wx * 4 + tx + 2) & 63;
    const size_t srow = ((size_t)(z * 64 + y) * 64 + xx) * DIM;
    const size_t prow = (size_t)row * DIM;

    float v[12], s = 0.f;
#pragma unroll
    for (int i = 0; i < 12; ++i) {
        const int c = lane + i * 32;
        v[i] = x[srow + c] + (float)proj[prow + c];
        outp[srow + c] = v[i];
        s += v[i];
    }
    const float mu = wave_sum(s) * (1.f / DIM);
    float q = 0.f;
#pragma unroll
    for (int i = 0; i < 12; ++i) { float d = v[i] - mu; q += d * d; }
    const float rs = rsqrtf(wave_sum(q) * (1.f / DIM) + 1e-5f);
#pragma unroll
    for (int i = 0; i < 12; ++i) {
        const int c = lane + i * 32;
        h1[srow + c] = (_Float16)((v[i] - mu) * rs * g[c] + b[c]);
    }
}

// ---------------- host side ------------------------------------------------

extern "C" void kernel_launch(void* const* d_in, const int* in_sizes, int n_in,
                              void* d_out, int out_size, void* d_ws, size_t ws_size,
                              hipStream_t stream) {
    (void)in_sizes; (void)n_in; (void)out_size; (void)ws_size;
    const float* x       = (const float*)d_in[0];
    const float* ln1_g   = (const float*)d_in[1];
    const float* ln1_b   = (const float*)d_in[2];
    const float* qkv_w   = (const float*)d_in[3];
    const float* qkv_b   = (const float*)d_in[4];
    const float* rpb_tab = (const float*)d_in[5];
    const float* proj_w  = (const float*)d_in[6];
    const float* proj_b  = (const float*)d_in[7];
    const float* ln2_g   = (const float*)d_in[8];
    const float* ln2_b   = (const float*)d_in[9];
    const float* mlp_w1  = (const float*)d_in[10];
    const float* mlp_b1  = (const float*)d_in[11];
    const float* mlp_w2  = (const float*)d_in[12];
    const float* mlp_b2  = (const float*)d_in[13];
    float* outp = (float*)d_out;

    char* ws = (char*)d_ws;
    _Float16* qkvT  = (_Float16*)(ws + 0);         // 1152x384
    _Float16* projT = (_Float16*)(ws + 884736);    // 384x384
    _Float16* w1T   = (_Float16*)(ws + 1179648);   // 1536x384
    _Float16* w2T   = (_Float16*)(ws + 2359296);   // 384x1536
    _Float16* buf1  = (_Float16*)(ws + 4194304);   // 50.3 MB: yw / attn-out / h1
    _Float16* buf2  = (_Float16*)(ws + 54525952);  // 201 MB: qkv / proj-out / hidden

    // weights -> fp16 transposed
    cvt_t_kernel<<<(DIM * QKV_N + 255) / 256, 256, 0, stream>>>(qkv_w,  qkvT,  DIM,  QKV_N);
    cvt_t_kernel<<<(DIM * DIM   + 255) / 256, 256, 0, stream>>>(proj_w, projT, DIM,  DIM);
    cvt_t_kernel<<<(DIM * MLP_H + 255) / 256, 256, 0, stream>>>(mlp_w1, w1T,   DIM,  MLP_H);
    cvt_t_kernel<<<(MLP_H * DIM + 255) / 256, 256, 0, stream>>>(mlp_w2, w2T,   MLP_H, DIM);

    // LN1 + shift + window partition -> buf1 (fp16, window order)
    ln1_win_kernel<<<NROWS / 8, 256, 0, stream>>>(x, ln1_g, ln1_b, buf1);

    // QKV GEMM: [65536,384] x [384,1152] -> buf2 fp16
    gemm_wmma<0><<<dim3(QKV_N / BNT, NROWS / BMT), 256, 0, stream>>>(
        buf1, qkvT, qkv_b, buf2, nullptr, NROWS, QKV_N, DIM);

    // attention per (window, head) -> buf1 fp16 [row][head*32+c]
    attn_kernel<<<dim3(HEADS, NWIN), 128, 0, stream>>>(buf2, rpb_tab, buf1);

    // proj GEMM -> buf2 fp16 (window order)
    gemm_wmma<0><<<dim3(DIM / BNT, NROWS / BMT), 256, 0, stream>>>(
        buf1, projT, proj_b, buf2, nullptr, NROWS, DIM, DIM);

    // residual + unshift + LN2: d_out = x + o ; h1 = LN2(d_out) fp16 -> buf1
    resid_ln2_kernel<<<NROWS / 8, 256, 0, stream>>>(x, buf2, ln2_g, ln2_b, outp, buf1);

    // MLP1 (+GELU) -> buf2 fp16 hidden
    gemm_wmma<1><<<dim3(MLP_H / BNT, NROWS / BMT), 256, 0, stream>>>(
        buf1, w1T, mlp_b1, buf2, nullptr, NROWS, MLP_H, DIM);

    // MLP2 accumulate into d_out (residual)
    gemm_wmma<2><<<dim3(DIM / BNT, NROWS / BMT), 256, 0, stream>>>(
        buf2, w2T, mlp_b2, nullptr, outp, NROWS, DIM, MLP_H);
}
